// Attention_60859686584336
// MI455X (gfx1250) — matrix operand order repaired and checked by reading, not compiled, not run on hardware
//
#include <hip/hip_runtime.h>
#include <math.h>

#define S_LEN 2048
#define B_N   64
#define E_N   512
#define D_N   512

typedef __bf16        v16bf  __attribute__((ext_vector_type(16)));
typedef __bf16        bf16x2 __attribute__((ext_vector_type(2)));
typedef float         v8f    __attribute__((ext_vector_type(8)));
typedef unsigned int  v8u    __attribute__((ext_vector_type(8)));

union ABCast { v8u u; v16bf bf; };

// f32 pair -> packed bf16 dword (RNE). Plain casts: the backend selects the
// hardware packed convert (v_cvt_pk_bf16_f32) when available.
__device__ __forceinline__ unsigned int pk_bf16(float a, float b) {
  bf16x2 p = { static_cast<__bf16>(a), static_cast<__bf16>(b) };
  return __builtin_bit_cast(unsigned int, p);
}

__device__ __forceinline__ float fast_tanh(float x) {
#if __has_builtin(__builtin_amdgcn_tanhf)
  return __builtin_amdgcn_tanhf(x);      // v_tanh_f32 (gfx1250 trans op)
#else
  return tanhf(x);
#endif
}

// -------------------------------------------------------------------------
// Kernel 1: W_enc (f32 [E,E], row f, contiguous e) -> bf16 pre-swizzled into
// the WMMA B-matrix (32x16, K=e, N=f) lane layout.
// wsB layout: [ftile(32)][ktile(16)][lane(32)][j(8 packed dwords)]
// lane: n = lane&15, kbase = (lane>>4)*16 ; dword j holds K = kbase+2j, +1
// -------------------------------------------------------------------------
__global__ void k_pack_B(const float* __restrict__ W_enc,
                         unsigned int* __restrict__ wsB) {
  int tid  = blockIdx.x * blockDim.x + threadIdx.x;   // [0, 131072)
  int j    = tid & 7;
  int lane = (tid >> 3) & 31;
  int kt   = (tid >> 8) & 15;
  int ft   = tid >> 12;
  int n    = lane & 15;
  int kb   = (lane >> 4) * 16;
  int e    = kt * 32 + kb + 2 * j;
  int f    = ft * 16 + n;
  wsB[tid] = pk_bf16(W_enc[f * E_N + e], W_enc[f * E_N + e + 1]);
}

// -------------------------------------------------------------------------
// Kernel 2: addvec[b,f] = b_enc[f] + b_dec[f] + sum_d dec[b,d]*W_dec[f,d]
// -------------------------------------------------------------------------
__global__ void k_addvec(const float* __restrict__ dec,
                         const float* __restrict__ Wd,
                         const float* __restrict__ be,
                         const float* __restrict__ bd,
                         float* __restrict__ out) {
  int idx = blockIdx.x * 256 + threadIdx.x;           // b*512 + f
  int b = idx >> 9, f = idx & 511;
  const float4* dv = (const float4*)(dec + b * D_N);
  const float4* wr = (const float4*)(Wd + (size_t)f * D_N);
  float acc = 0.f;
  for (int d = 0; d < D_N / 4; ++d) {
    float4 x = dv[d], y = wr[d];
    acc += x.x * y.x + x.y * y.y + x.z * y.z + x.w * y.w;
  }
  out[idx] = acc + be[f] + bd[f];
}

// -------------------------------------------------------------------------
// Kernel 3: fused GEMM + tanh + scale-dot -> raw scores[b, s]
// Grid (S/128, B); 512 threads = 16 waves.
// Workgroup tile: M=128 (8 m-tiles), N=512 (wave w owns N-tiles 2w, 2w+1).
// A (enc rows) is converted f32->bf16 cooperatively into LDS in the WMMA
// A-fragment layout, staged in four K-quarters of 4 k-tiles each (32 KB).
// Software-pipelined: A double-buffered across m-tiles (ds), B
// double-buffered across k-tiles (global).
// -------------------------------------------------------------------------
__global__ __launch_bounds__(512)
void k_attn_scores(const float* __restrict__ enc,
                   const unsigned int* __restrict__ wsB,
                   const float* __restrict__ addvec,
                   const float* __restrict__ scale,
                   float* __restrict__ scores) {
  // [mt(8)][kt4(4)][lane(32)][j(8)] packed-bf16 dwords = 32 KB
  __shared__ alignas(32) unsigned int ldsA[8 * 4 * 32 * 8];
  __shared__ float sm[16][8][16];                      // 8 KB

  const int sblk = blockIdx.x;        // 128-row stripe index
  const int b    = blockIdx.y;
  const int tid  = threadIdx.x;
  const int lane = tid & 31;
  const int wave = tid >> 5;
  const int hi   = lane >> 4;
  const int mn   = lane & 15;

  v8f zero = {0.f, 0.f, 0.f, 0.f, 0.f, 0.f, 0.f, 0.f};
  v8f acc[16];                        // [mt][t] = acc[mt*2+t]
#pragma unroll
  for (int i = 0; i < 16; ++i) acc[i] = zero;

  const v8u* bptr = (const v8u*)wsB;
  const v8u* aptr = (const v8u*)ldsA;

  for (int stage = 0; stage < 4; ++stage) {
    // ---- stage: convert 128 rows x 128 e-columns into LDS A fragments ----
#pragma unroll 4
    for (int i = 0; i < 16; ++i) {
      int idx = tid + i * 512;                 // [0, 8192)
      int j   = idx & 7;
      int ln  = (idx >> 3) & 31;
      int kt4 = (idx >> 8) & 3;
      int mt  = idx >> 10;
      int row = mt * 16 + (ln & 15);
      int kb  = (ln >> 4) * 16;
      int k   = kb + 2 * j;
      int e   = (stage * 4 + kt4) * 32 + k;
      const float* src = enc + ((size_t)(sblk * 128 + row) * B_N + b) * E_N + e;
      ldsA[idx] = pk_bf16(src[0], src[1]);
    }
    __syncthreads();

    // ---- compute: 4 k-tiles; 2 B frags reused across 8 m-tiles ----
    // B double-buffered across k-tiles; A double-buffered across m-tiles.
    v8u Bf0 = bptr[((wave * 2 + 0) * 16 + stage * 4) * 32 + lane];
    v8u Bf1 = bptr[((wave * 2 + 1) * 16 + stage * 4) * 32 + lane];
#pragma unroll
    for (int kt4 = 0; kt4 < 4; ++kt4) {
      v8u Bn0, Bn1;
      if (kt4 < 3) {
        const int ktn = stage * 4 + kt4 + 1;
        Bn0 = bptr[((wave * 2 + 0) * 16 + ktn) * 32 + lane];
        Bn1 = bptr[((wave * 2 + 1) * 16 + ktn) * 32 + lane];
      }
      ABCast Ac, An, B0c, B1c;
      B0c.u = Bf0;
      B1c.u = Bf1;
      Ac.u = aptr[(0 * 4 + kt4) * 32 + lane];
#pragma unroll
      for (int mt = 0; mt < 8; ++mt) {
        if (mt < 7) An.u = aptr[((mt + 1) * 4 + kt4) * 32 + lane];
        acc[mt * 2 + 0] = __builtin_amdgcn_wmma_f32_16x16x32_bf16(
            false, Ac.bf, false, B0c.bf, (short)0, acc[mt * 2 + 0], false, false);
        acc[mt * 2 + 1] = __builtin_amdgcn_wmma_f32_16x16x32_bf16(
            false, Ac.bf, false, B1c.bf, (short)0, acc[mt * 2 + 1], false, false);
        if (mt < 7) Ac = An;
      }
      if (kt4 < 3) { Bf0 = Bn0; Bf1 = Bn1; }
    }
    __syncthreads();   // safe to overwrite ldsA with next stage
  }

  // ---- epilogue: score[m] = sum_f scale[f]*tanh(C[m,f] + addvec[b,f]) ----
  // C element (v, lane) of tile (mt, t): local row = mt*16 + v + 8*hi,
  // column f = (wave*2 + t)*16 + (lane&15).
  float part[8][8];
#pragma unroll
  for (int mt = 0; mt < 8; ++mt)
#pragma unroll
    for (int v = 0; v < 8; ++v) part[mt][v] = 0.f;

#pragma unroll
  for (int t = 0; t < 2; ++t) {
    int f = (wave * 2 + t) * 16 + mn;
    float av = addvec[b * E_N + f];
    float sc = scale[f];
#pragma unroll
    for (int mt = 0; mt < 8; ++mt)
#pragma unroll
      for (int v = 0; v < 8; ++v)
        part[mt][v] += sc * fast_tanh(acc[mt * 2 + t][v] + av);
  }

  // reduce over the 16 lanes of each half-wave (sums 16 N columns per tile)
#pragma unroll
  for (int off = 1; off < 16; off <<= 1)
#pragma unroll
    for (int mt = 0; mt < 8; ++mt)
#pragma unroll
      for (int v = 0; v < 8; ++v)
        part[mt][v] += __shfl_xor(part[mt][v], off, 32);

  if (mn == 0) {
#pragma unroll
    for (int mt = 0; mt < 8; ++mt)
#pragma unroll
      for (int v = 0; v < 8; ++v)
        sm[wave][mt][v + 8 * hi] = part[mt][v];
  }
  __syncthreads();

  if (tid < 128) {
    float s = 0.f;
#pragma unroll
    for (int w = 0; w < 16; ++w) s += sm[w][tid >> 4][tid & 15];  // fixed order
    scores[b * S_LEN + sblk * 128 + tid] = s;
  }
}

// -------------------------------------------------------------------------
// Kernel 4: masked softmax over S per batch. 256 threads, 8 elems/thread.
// -------------------------------------------------------------------------
__global__ __launch_bounds__(256)
void k_softmax(const float* __restrict__ scores,
               const int* __restrict__ lens,
               float* __restrict__ wout) {
  const int b   = blockIdx.x;
  const int tid = threadIdx.x;
  const int len = lens[b];
  __shared__ float red[256];

  float v[8];
  float mx = -3.4e38f;
#pragma unroll
  for (int i = 0; i < 8; ++i) {
    int s = tid + i * 256;
    float x = scores[b * S_LEN + s];
    x = (s < len) ? x : -10000000000.0f;
    v[i] = x;
    mx = fmaxf(mx, x);
  }
  red[tid] = mx;
  __syncthreads();
  for (int o = 128; o > 0; o >>= 1) {
    if (tid < o) red[tid] = fmaxf(red[tid], red[tid + o]);
    __syncthreads();
  }
  mx = red[0];
  __syncthreads();

  float sum = 0.f;
#pragma unroll
  for (int i = 0; i < 8; ++i) {
    v[i] = expf(v[i] - mx);
    sum += v[i];
  }
  red[tid] = sum;
  __syncthreads();
  for (int o = 128; o > 0; o >>= 1) {
    if (tid < o) red[tid] += red[tid + o];
    __syncthreads();
  }
  float inv = 1.0f / red[0];
#pragma unroll
  for (int i = 0; i < 8; ++i)
    wout[b * S_LEN + tid + i * 256] = v[i] * inv;
}

// -------------------------------------------------------------------------
// Kernel 5: partial context sums over 128-s chunks (deterministic, no atomics)
// part[ch][b][e] = sum_{s in chunk} w[b,s] * enc[s,b,e]
// -------------------------------------------------------------------------
__global__ __launch_bounds__(512)
void k_outpart(const float* __restrict__ enc,
               const float* __restrict__ w,
               float* __restrict__ part) {
  const int ch = blockIdx.x, b = blockIdx.y, e = threadIdx.x;
  const int s0 = ch * 128;
  float acc = 0.f;
  for (int i = 0; i < 128; ++i) {
    float wv = w[b * S_LEN + s0 + i];
    acc += wv * enc[((size_t)(s0 + i) * B_N + b) * E_N + e];
  }
  part[((size_t)ch * B_N + b) * E_N + e] = acc;
}

// Kernel 6: reduce the 16 chunk partials, fixed order
__global__ void k_outreduce(const float* __restrict__ part,
                            float* __restrict__ out) {
  int idx = blockIdx.x * 256 + threadIdx.x;   // [0, B*E)
  float s = 0.f;
#pragma unroll
  for (int ch = 0; ch < 16; ++ch) s += part[ch * (B_N * E_N) + idx];
  out[idx] = s;
}

// -------------------------------------------------------------------------
extern "C" void kernel_launch(void* const* d_in, const int* in_sizes, int n_in,
                              void* d_out, int out_size, void* d_ws, size_t ws_size,
                              hipStream_t stream) {
  const float* enc   = (const float*)d_in[0];   // [S,B,E]
  const int*   lens  = (const int*)  d_in[1];   // [B]
  const float* dec   = (const float*)d_in[2];   // [1,B,D]
  const float* W_enc = (const float*)d_in[3];   // [E,E]
  const float* b_enc = (const float*)d_in[4];   // [E]
  const float* W_dec = (const float*)d_in[5];   // [E,D]
  const float* b_dec = (const float*)d_in[6];   // [E]
  const float* scale = (const float*)d_in[7];   // [E]

  float* out_ctx = (float*)d_out;               // [1,B,E] = 32768 floats
  float* out_w   = (float*)d_out + B_N * E_N;   // [B,1,S] = 131072 floats

  // workspace carve-up (dwords)
  unsigned int* wsB    = (unsigned int*)d_ws;                         // 131072 u32
  float*        addvec = (float*)d_ws + 131072;                       // 32768 f
  float*        scores = (float*)d_ws + 131072 + 32768;               // 131072 f
  float*        part   = (float*)d_ws + 131072 + 32768 + 131072;      // 524288 f

  k_pack_B     <<<512, 256, 0, stream>>>(W_enc, wsB);
  k_addvec     <<<128, 256, 0, stream>>>(dec, W_dec, b_enc, b_dec, addvec);
  k_attn_scores<<<dim3(S_LEN / 128, B_N), 512, 0, stream>>>(enc, wsB, addvec, scale, scores);
  k_softmax    <<<B_N, 256, 0, stream>>>(scores, lens, out_w);
  k_outpart    <<<dim3(16, B_N), 512, 0, stream>>>(enc, out_w, part);
  k_outreduce  <<<128, 256, 0, stream>>>(part, out_ctx);
}
